// KANLinear_45586782880324
// MI455X (gfx1250) — compile-verified, matrix operand order checked by reading, not run on hardware
//
#include <hip/hip_runtime.h>
#include <hip/hip_bf16.h>

// ---------------- problem constants (from reference) ----------------
#define BATCH   8192
#define IN_F    1024
#define OUT_F   1024
#define KCOMP   4                 // silu + 3 non-zero spline basis columns
#define KTOT    (IN_F * KCOMP)    // 4096 augmented K
#define BKC     32                // K per chunk (bf16 WMMA K)
#define NCHUNK  (KTOT / BKC)      // 128
#define BM      128               // batch rows per block
#define BN      128               // out features per block

typedef __bf16 v16bf __attribute__((ext_vector_type(16)));
typedef float  v8f   __attribute__((ext_vector_type(8)));
typedef unsigned int       u32x4 __attribute__((ext_vector_type(4)));
typedef int                i32x4 __attribute__((ext_vector_type(4)));
typedef int                i32x8 __attribute__((ext_vector_type(8)));
typedef unsigned long long u64x4 __attribute__((ext_vector_type(4)));

#if __has_include(<hip/amd_detail/amd_gfx1250_TDM.h>)
#define KAN_TDM_6ARG 1   // therock headers -> 6-arg tensor builtin
#endif

static __device__ __forceinline__ unsigned short f2bf_bits(float f) {
  union { float f; unsigned u; } v; v.f = f;
  unsigned r = v.u + 0x7FFFu + ((v.u >> 16) & 1u);  // round-to-nearest-even
  return (unsigned short)(r >> 16);
}

// Faithful b_splines recursion (constant grid folds at compile time) + silu.
// Only basis columns 0..2 are ever non-zero given the reference's guards.
static __device__ __forceinline__ void kan_feats(float x, float out[4]) {
  const float EPSc = 1e-8f;
  float b[8];
#pragma unroll
  for (int i = 0; i < 8; ++i) {
    float gi  = -1.0f + 0.4f * (float)i;
    float gi1 = -1.0f + 0.4f * (float)(i + 1);
    b[i] = (i < 5 && x >= gi && x < gi1) ? 1.0f : 0.0f;
  }
#pragma unroll
  for (int k = 1; k <= 3; ++k) {
    float nb[8];
#pragma unroll
    for (int i = 0; i < 8; ++i) {
      float t = 0.0f;
      if (i < 8 - k && i + k < 6) {
        float gi  = -1.0f + 0.4f * (float)i;
        float gik = -1.0f + 0.4f * (float)(i + k);
        t = ((x - gi) / (gik - gi + EPSc)) * b[i];
        if (i + k + 1 < 6) {
          float gik1 = -1.0f + 0.4f * (float)(i + k + 1);
          float gi1  = -1.0f + 0.4f * (float)(i + 1);
          t += ((gik1 - x) / (gik1 - gi1 + EPSc)) * b[i + 1];
        }
      }
      nb[i] = t;
    }
#pragma unroll
    for (int i = 0; i < 8; ++i) b[i] = nb[i];
  }
  float sig = 1.0f / (1.0f + __expf(-x));
  out[0] = x * sig;   // silu
  out[1] = b[0];
  out[2] = b[1];
  out[3] = b[2];
}

// TDM 1D copy: 8KB (4096 x 2B) contiguous global -> LDS byte offset.
// Descriptor packing per CDNA5 ISA (groups 0/1; 2/3 zero).
static __device__ __forceinline__ void tdm_load_8kb(unsigned lds_byte_off,
                                                    const void* gptr) {
#if __has_builtin(__builtin_amdgcn_tensor_load_to_lds)
  unsigned long long ga = (unsigned long long)(size_t)gptr;
  u32x4 g0;
  g0[0] = 1u;                                   // count=1, user-mode load
  g0[1] = lds_byte_off;                         // lds_addr [63:32]
  g0[2] = (unsigned)(ga & 0xFFFFFFFFu);         // global_addr [95:64]
  g0[3] = (unsigned)((ga >> 32) & 0x01FFFFFFu)  // global_addr [120:96]
        | (2u << 30);                           // type=2 ("image")
  i32x8 g1;
  g1[0] = 0x00010000;                  // data_size=1 (2 bytes), no flags
  g1[1] = (int)(4096u << 16);          // tensor_dim0 lo16 (=4096) at [63:48]
  g1[2] = (int)((4096u >> 16) | (1u << 16)); // tensor_dim0 hi | tensor_dim1=1
  g1[3] = (int)(4096u << 16);          // tensor_dim1 hi=0 | tile_dim0=4096
  g1[4] = 0;                           // tile_dim1=0, tile_dim2=0 (1D tile)
  g1[5] = 4096;                        // tensor_dim0_stride lo
  g1[6] = 0;                           // stride hi | dim1_stride lo
  g1[7] = 0;
  i32x4 z4 = {0, 0, 0, 0};
#ifdef KAN_TDM_6ARG
  i32x8 z8 = {0, 0, 0, 0, 0, 0, 0, 0};
  __builtin_amdgcn_tensor_load_to_lds(g0, g1, z4, z4, z8, 0);
#else
  __builtin_amdgcn_tensor_load_to_lds(g0, g1, z4, z4, 0);
#endif
#else
  (void)lds_byte_off; (void)gptr;
#endif
}

// ---------------- kernel 1: per-output-row scaler sums ----------------
__global__ void kan_scale_sum(const float* __restrict__ scaler,
                              float* __restrict__ ssum) {
  __shared__ float red[256];
  int o = blockIdx.x;
  float s = 0.0f;
  for (int i = threadIdx.x; i < IN_F; i += 256) s += scaler[o * IN_F + i];
  red[threadIdx.x] = s;
  __syncthreads();
#pragma unroll
  for (int w = 128; w > 0; w >>= 1) {
    if (threadIdx.x < (unsigned)w) red[threadIdx.x] += red[threadIdx.x + w];
    __syncthreads();
  }
  if (threadIdx.x == 0) ssum[o] = red[0];
}

// ---------------- kernel 2: pack augmented weights to bf16 in WMMA-B
// fragment order: wpack[kc][o_subtile(64)][lane(32)][elem(16)] ----------------
__global__ void kan_wpack(const float* __restrict__ baseW,
                          const float* __restrict__ splineW,
                          const float* __restrict__ ssum,
                          unsigned short* __restrict__ wpack) {
  int d = blockIdx.x * 256 + threadIdx.x;       // 0 .. 4M-1
  int e    = d & 15;
  int lane = (d >> 4) & 31;
  int sub  = (d >> 9) & 63;
  int kc   = d >> 15;
  int o    = sub * 16 + (lane & 15);
  int kloc = (e & 7) + ((lane >> 4) * 8) + ((e >= 8) ? 16 : 0);
  int gk   = kc * BKC + kloc;
  int i    = gk >> 2;
  int c    = gk & 3;
  float v  = (c == 0) ? baseW[o * IN_F + i]
                      : ssum[o] * splineW[(o * IN_F + i) * 8 + (c - 1)];
  wpack[d] = f2bf_bits(v);
}

// ---------------- kernel 3: pack augmented activations (silu + basis) to
// bf16 in WMMA-A fragment order: apack[kc][m_subtile(512)][lane(32)][elem(16)].
// One-time VALU cost; removes all basis math from the GEMM hot loop. --------
__global__ void kan_apack(const float* __restrict__ X,
                          unsigned short* __restrict__ apack) {
  int t = blockIdx.x * 256 + threadIdx.x;  // 0 .. 2,097,151  (kc, sub, lane)
  int lane = t & 31;
  int sub  = (t >> 5) & 511;
  int kc   = t >> 14;
  int row      = sub * 16 + (lane & 15);
  int laneHalf = lane >> 4;

  unsigned long long q[4];
#pragma unroll
  for (int eg = 0; eg < 4; ++eg) {
    // fragment element-group eg holds feature f = (eg&1) | laneHalf<<1 | (eg>>1)<<2
    int f = (eg & 1) | (laneHalf << 1) | ((eg >> 1) << 2);
    float xv = X[(long)row * IN_F + kc * 8 + f];
    float ft[4];
    kan_feats(xv, ft);
    q[eg] = (unsigned long long)f2bf_bits(ft[0]) |
            ((unsigned long long)f2bf_bits(ft[1]) << 16) |
            ((unsigned long long)f2bf_bits(ft[2]) << 32) |
            ((unsigned long long)f2bf_bits(ft[3]) << 48);
  }
  u64x4 pk; pk[0] = q[0]; pk[1] = q[1]; pk[2] = q[2]; pk[3] = q[3];
  *(u64x4*)&apack[(size_t)t << 4] = pk;   // coalesced 32B store
}

// ---------------- kernel 4: WMMA GEMM, TDM double-buffered tiles ----------
// Dynamic LDS only => descriptor lds_addr base is offset 0.
// Layout (bytes): buf p in [p*16384, p*16384+16384): A tile 8KB then B tile 8KB.
extern __shared__ unsigned short kan_smem[];

__global__ __launch_bounds__(256) void kan_gemm(
    const unsigned short* __restrict__ apack,
    const unsigned short* __restrict__ wpack,
    float* __restrict__ out) {
  const int tid   = threadIdx.x;
  const int lane  = tid & 31;
  const int wave  = tid >> 5;
  const int waveM = wave >> 2;        // 0..1  (4 M-subtiles each)
  const int waveN = wave & 3;         // 0..3  (2 N-subtiles each)
  const int bN    = blockIdx.x;       // 0..7
  const int bM    = blockIdx.y;       // 0..63

  v8f acc[4][2];
#pragma unroll
  for (int mt = 0; mt < 4; ++mt)
#pragma unroll
    for (int nt = 0; nt < 2; ++nt) acc[mt][nt] = (v8f)0.0f;

#if __has_builtin(__builtin_amdgcn_tensor_load_to_lds)
  // ---- prologue: TDM chunk 0 into buffer 0 (wave 0 issues for the block) ----
  if (wave == 0) {
    tdm_load_8kb(0u,    apack + ((size_t)(0 * 512 + bM * 8) << 9));
    tdm_load_8kb(8192u, wpack + ((size_t)(0 * 64  + bN * 8) << 9));
  }
  for (int kc = 0; kc < NCHUNK; ++kc) {
    const int p = kc & 1;
    if (wave == 0) __builtin_amdgcn_s_wait_tensorcnt(0);
    __syncthreads();                       // tile kc visible to all waves
    if (wave == 0 && kc + 1 < NCHUNK) {    // prefetch kc+1 into other buffer
      tdm_load_8kb((unsigned)((p ^ 1) * 16384),
                   apack + ((size_t)((kc + 1) * 512 + bM * 8) << 9));
      tdm_load_8kb((unsigned)((p ^ 1) * 16384 + 8192),
                   wpack + ((size_t)((kc + 1) * 64 + bN * 8) << 9));
    }
    const int aoff = p * 8192;             // ushort offsets into LDS
    const int boff = p * 8192 + 4096;
    v16bf afrag[4];
#pragma unroll
    for (int mt = 0; mt < 4; ++mt)
      afrag[mt] = *(const v16bf*)&kan_smem[aoff + ((waveM * 4 + mt) << 9) + (lane << 4)];
    v16bf bfrag[2];
#pragma unroll
    for (int nt = 0; nt < 2; ++nt)
      bfrag[nt] = *(const v16bf*)&kan_smem[boff + ((waveN * 2 + nt) << 9) + (lane << 4)];
#pragma unroll
    for (int mt = 0; mt < 4; ++mt)
#pragma unroll
      for (int nt = 0; nt < 2; ++nt)
        acc[mt][nt] = __builtin_amdgcn_wmma_f32_16x16x32_bf16(
            false, afrag[mt], false, bfrag[nt], (short)0, acc[mt][nt],
            false, false);
    __syncthreads();                       // all reads of buf p done
  }
#else
  // ---- fallback: register-staged double buffer (no TDM builtin) ----
  {
    v16bf a0 = *(const v16bf*)(apack + ((size_t)(bM * 8) << 9) + (tid << 4));
    v16bf b0 = *(const v16bf*)(wpack + ((size_t)(bN * 8) << 9) + (tid << 4));
    *(v16bf*)&kan_smem[tid << 4] = a0;
    *(v16bf*)&kan_smem[4096 + (tid << 4)] = b0;
  }
  for (int kc = 0; kc < NCHUNK; ++kc) {
    const int p = kc & 1;
    __syncthreads();
    if (kc + 1 < NCHUNK) {
      v16bf a = *(const v16bf*)(apack + ((size_t)((kc + 1) * 512 + bM * 8) << 9) + (tid << 4));
      v16bf b = *(const v16bf*)(wpack + ((size_t)((kc + 1) * 64  + bN * 8) << 9) + (tid << 4));
      *(v16bf*)&kan_smem[(p ^ 1) * 8192 + (tid << 4)] = a;
      *(v16bf*)&kan_smem[(p ^ 1) * 8192 + 4096 + (tid << 4)] = b;
    }
    const int aoff = p * 8192;
    const int boff = p * 8192 + 4096;
    v16bf afrag[4];
#pragma unroll
    for (int mt = 0; mt < 4; ++mt)
      afrag[mt] = *(const v16bf*)&kan_smem[aoff + ((waveM * 4 + mt) << 9) + (lane << 4)];
    v16bf bfrag[2];
#pragma unroll
    for (int nt = 0; nt < 2; ++nt)
      bfrag[nt] = *(const v16bf*)&kan_smem[boff + ((waveN * 2 + nt) << 9) + (lane << 4)];
#pragma unroll
    for (int mt = 0; mt < 4; ++mt)
#pragma unroll
      for (int nt = 0; nt < 2; ++nt)
        acc[mt][nt] = __builtin_amdgcn_wmma_f32_16x16x32_bf16(
            false, afrag[mt], false, bfrag[nt], (short)0, acc[mt][nt],
            false, false);
    __syncthreads();
  }
#endif

  // ---- epilogue: f32 store, C-layout M = r + (lane>=16)*8, N = lane&15 ----
  const int mbase = bM * BM + waveM * 64 + ((lane >> 4) * 8);
  const int nbase = bN * BN + waveN * 32 + (lane & 15);
#pragma unroll
  for (int mt = 0; mt < 4; ++mt) {
#pragma unroll
    for (int nt = 0; nt < 2; ++nt) {
      int gm = mbase + mt * 16;
      int gn = nbase + nt * 16;
#pragma unroll
      for (int r = 0; r < 8; ++r)
        out[(long)(gm + r) * OUT_F + gn] = acc[mt][nt][r];
    }
  }
}

// ---------------- launcher ----------------
extern "C" void kernel_launch(void* const* d_in, const int* in_sizes, int n_in,
                              void* d_out, int out_size, void* d_ws, size_t ws_size,
                              hipStream_t stream) {
  const float* x             = (const float*)d_in[0];
  const float* base_weight   = (const float*)d_in[1];
  const float* spline_weight = (const float*)d_in[2];
  const float* spline_scaler = (const float*)d_in[3];
  // d_in[4] = grid: constants baked into kan_feats (linspace(-1,1,6))

  // workspace layout: ssum 4KB | wpack 8MB | apack 64MB   (~72MB total)
  float* ssum           = (float*)d_ws;
  unsigned short* wpack = (unsigned short*)((char*)d_ws + 4096);
  unsigned short* apack = (unsigned short*)((char*)d_ws + 4096 +
                                            (size_t)OUT_F * KTOT * 2);

  kan_scale_sum<<<OUT_F, 256, 0, stream>>>(spline_scaler, ssum);
  kan_wpack<<<(NCHUNK * 64 * 32 * 16) / 256, 256, 0, stream>>>(
      base_weight, spline_weight, ssum, wpack);
  kan_apack<<<(NCHUNK * 512 * 32) / 256, 256, 0, stream>>>(x, apack);
  kan_gemm<<<dim3(OUT_F / BN, BATCH / BM), 256, 32768, stream>>>(
      apack, wpack, (float*)d_out);
}
